// TemporalTransformer_57200374448660
// MI455X (gfx1250) — compile-verified, matrix operand order checked by reading, not run on hardware
//
#include <hip/hip_runtime.h>
#include <hip/hip_bf16.h>

// ---------------------------------------------------------------------------
// Temporal transformer on MI455X (gfx1250).
// All matmuls via v_wmma_f32_16x16x32_f16 (fp32 accumulate).
// fp32 master activations; weights pre-converted once per launch into
// transposed f16 [N][K] so every LDS stage is vectorized b128 traffic.
// ---------------------------------------------------------------------------

typedef __attribute__((ext_vector_type(16))) _Float16 v16h;
typedef __attribute__((ext_vector_type(8)))  _Float16 h8;
typedef __attribute__((ext_vector_type(8)))  float    v8f;

#define LDSP 40   // padded LDS row stride in halfs (80B) -> conflict-free b128 reads

enum { MODE_LIN = 0, MODE_CONV = 1, MODE_SCORES = 2, MODE_PV = 3, MODE_VPROJ = 4 };

struct GArgs {
  const float*    A;  int lda;     // fp32 activations, row-major
  const _Float16* Bh; int ldb;     // f16 transposed weights [N][K] (row stride ldb)
  const float*    Bf;              // fp32 B operand (SCORES only)
  const float*    bias;            // length N or null
  const float*    resid;           // [M,ldc] or null
  float*          C;  int ldc;
  _Float16*       Cv;              // VPROJ: f16 V^T out [B][D][L]
  int M, N, K;
  int sectionSize;                 // == K normally, == D for conv
  int shiftBase;                   // -2 for conv taps, 0 otherwise
  int Lseq;
  float scale;                     // scores scale
  int relu;
  int H, hd, D, Lq;
};

__device__ inline h8 hzero() {
  h8 r;
#pragma unroll
  for (int i = 0; i < 8; ++i) r[i] = (_Float16)0.f;
  return r;
}

__device__ inline h8 pack8(float4 a, float4 b) {
  h8 r;
  r[0] = (_Float16)a.x; r[1] = (_Float16)a.y; r[2] = (_Float16)a.z; r[3] = (_Float16)a.w;
  r[4] = (_Float16)b.x; r[5] = (_Float16)b.y; r[6] = (_Float16)b.z; r[7] = (_Float16)b.w;
  return r;
}

__device__ inline v16h frag_from_lds(const _Float16* base, int lane) {
  // ISA 16-bit operand layout: lane r=l&15 is the row/col index; the hi
  // half-wave holds K = hi*8..+7 (elems 0..7) and K = 16+hi*8..+7 (elems 8..15).
  const int r = lane & 15, hi = lane >> 4;
  const h8* p0 = (const h8*)(base + r * LDSP + hi * 8);
  const h8* p1 = (const h8*)(base + r * LDSP + 16 + hi * 8);
  h8 x = *p0, y = *p1;
  v16h a;
#pragma unroll
  for (int i = 0; i < 8; ++i) { a[i] = x[i]; a[i + 8] = y[i]; }
  return a;
}

// Block tile 64(M) x 128(N), BK=32. 4 waves in 2x2; each wave 32x64 = 8 WMMAs.
template <int MODE>
__global__ __launch_bounds__(128) void gemm_wmma(GArgs g) {
  constexpr bool F16B = (MODE != MODE_SCORES);
  __shared__ __attribute__((aligned(16))) _Float16 As[64 * LDSP];
  __shared__ __attribute__((aligned(16))) _Float16 Bs[128 * LDSP];

  const int tid  = threadIdx.x;
  const int lane = tid & 31;
  const int w    = tid >> 5;
  const int wr   = w >> 1, wc = w & 1;
  const int z    = blockIdx.z;

  const float*    Az = g.A;
  const _Float16* Bh = g.Bh;
  const float*    Bf = g.Bf;
  float*          Cz = g.C;
  int             ldb = g.ldb;
  int bz = 0, hh = 0;
  if (MODE == MODE_SCORES || MODE == MODE_PV) {
    bz = z / g.H; hh = z - bz * g.H;
    if (MODE == MODE_SCORES) {
      Az  = g.A + (size_t)bz * g.Lq * g.lda + (size_t)hh * g.hd;          // Q rows
      Bf  = g.Bf + (size_t)bz * g.Lq * g.D + (size_t)hh * g.hd;          // K rows
      ldb = g.D;
      Cz  = g.C + (size_t)z * g.Lq * g.Lq;
    } else {
      Az  = g.A + (size_t)z * g.Lq * g.Lq;                                // probs
      Bh  = g.Bh + ((size_t)bz * g.D + (size_t)hh * g.hd) * g.Lq;        // V^T rows
      ldb = g.Lq;
      Cz  = g.C + (size_t)bz * g.Lq * g.ldc + (size_t)hh * g.hd;
    }
  }

  const int m0 = blockIdx.y * 64;
  const int n0 = blockIdx.x * 128;
  const int mB = m0 / g.Lseq;                 // block never straddles a batch

  v8f acc[2][4] = {};

  int k0 = 0;
  for (int sec = 0; k0 < g.K; ++sec) {
    const int shift   = g.shiftBase + sec;    // conv causal tap shift
    const int secBase = sec * g.sectionSize;
    const int kend    = (secBase + g.sectionSize < g.K) ? (secBase + g.sectionSize) : g.K;
    for (; k0 < kend; k0 += 32) {
      // ---- stage A tile 64x32 (fp32 -> f16), 2 threads per row ----
      {
        const int row  = tid >> 1;
        const int c0   = (tid & 1) * 16;
        const int mg   = m0 + row;
        const int tpos = mg - mB * g.Lseq;
        h8 lo = hzero(), hi8 = hzero();
        if (tpos + shift >= 0 && tpos + shift < g.Lseq) {
          const float* src = Az + (size_t)(mg + shift) * g.lda + (k0 - secBase) + c0;
          const float4* s4 = (const float4*)src;
          lo  = pack8(s4[0], s4[1]);
          hi8 = pack8(s4[2], s4[3]);
          __builtin_prefetch(src + 32, 0, 3);
        }
        h8* dst = (h8*)&As[row * LDSP + c0];
        dst[0] = lo; dst[1] = hi8;
      }
      // ---- stage B tile 128x32 as Bs[n][k], one row per thread ----
      {
        const int ng = n0 + tid;
        h8 b0 = hzero(), b1 = hzero(), b2 = hzero(), b3 = hzero();
        if (ng < g.N) {
          if (F16B) {
            const h8* src = (const h8*)(Bh + (size_t)ng * ldb + k0);
            b0 = src[0]; b1 = src[1]; b2 = src[2]; b3 = src[3];
            __builtin_prefetch((const void*)(src + 4), 0, 3);
          } else {
            const float4* s4 = (const float4*)(Bf + (size_t)ng * ldb + k0);
            b0 = pack8(s4[0], s4[1]); b1 = pack8(s4[2], s4[3]);
            b2 = pack8(s4[4], s4[5]); b3 = pack8(s4[6], s4[7]);
            __builtin_prefetch((const void*)(s4 + 8), 0, 3);
          }
        }
        h8* dst = (h8*)&Bs[tid * LDSP];
        dst[0] = b0; dst[1] = b1; dst[2] = b2; dst[3] = b3;
      }
      __syncthreads();

      v16h a[2], b[4];
      a[0] = frag_from_lds(&As[(wr * 32 + 0)  * LDSP], lane);
      a[1] = frag_from_lds(&As[(wr * 32 + 16) * LDSP], lane);
#pragma unroll
      for (int j = 0; j < 4; ++j)
        b[j] = frag_from_lds(&Bs[(wc * 64 + j * 16) * LDSP], lane);

#pragma unroll
      for (int ti = 0; ti < 2; ++ti)
#pragma unroll
        for (int tj = 0; tj < 4; ++tj)
          acc[ti][tj] = __builtin_amdgcn_wmma_f32_16x16x32_f16(
              false, a[ti], false, b[tj], (short)0, acc[ti][tj], false, false);
      __syncthreads();
    }
  }

  // ---- epilogue: C/D layout: lane l -> row (l>>4)*8 + i, col l&15 ----
  const int r15 = lane & 15, hi = lane >> 4;
#pragma unroll
  for (int ti = 0; ti < 2; ++ti)
#pragma unroll
    for (int tj = 0; tj < 4; ++tj) {
      const int nn = n0 + wc * 64 + tj * 16 + r15;
      if (nn >= g.N) continue;
#pragma unroll
      for (int i = 0; i < 8; ++i) {
        const int mm = m0 + wr * 32 + ti * 16 + hi * 8 + i;
        if (mm >= g.M) continue;
        float v = acc[ti][tj][i];
        if (MODE == MODE_SCORES) {
          v *= g.scale;
          if (nn > mm) v = -3.0e38f;          // causal mask
          Cz[(size_t)mm * g.ldc + nn] = v;
        } else if (MODE == MODE_VPROJ) {
          v += g.bias[nn];
          const int tpos = mm - mB * g.Lseq;  // V^T[b][n][t], f16
          g.Cv[((size_t)mB * g.D + nn) * g.Lseq + tpos] = (_Float16)v;
        } else {
          if (g.bias)  v += g.bias[nn];
          if (g.relu)  v = fmaxf(v, 0.f);
          if (g.resid) v += g.resid[(size_t)mm * g.ldc + nn];
          Cz[(size_t)mm * g.ldc + nn] = v;
        }
      }
    }
}

// ---------------------------------------------------------------------------
// Weight prep: fp32 [K][N] -> f16 [N][K] tiled transpose.
// ---------------------------------------------------------------------------
__global__ __launch_bounds__(256) void transpose_f16_kernel(
    const float* __restrict__ w, _Float16* __restrict__ wt, int K, int N) {
  __shared__ float t[32][33];
  const int tx = threadIdx.x & 31, ty = threadIdx.x >> 5;   // 32 x 8
  const int n0 = blockIdx.x * 32, k0 = blockIdx.y * 32;
#pragma unroll
  for (int j = 0; j < 32; j += 8)
    t[ty + j][tx] = w[(size_t)(k0 + ty + j) * N + n0 + tx];
  __syncthreads();
#pragma unroll
  for (int j = 0; j < 32; j += 8)
    wt[(size_t)(n0 + ty + j) * K + k0 + tx] = (_Float16)t[tx][ty + j];
}

// conv_w [Dout][Din][3] fp32 -> wt [Dout][tap*512+din] f16 (D = 512 fixed)
__global__ __launch_bounds__(256) void conv_wt_kernel(
    const float* __restrict__ cw, _Float16* __restrict__ wt) {
  const int idx = blockIdx.x * 256 + threadIdx.x;   // over 512*1536
  if (idx >= 512 * 1536) return;
  const int n = idx / 1536;
  const int k = idx - n * 1536;
  const int tap = k >> 9;
  const int din = k & 511;
  wt[idx] = (_Float16)cw[((size_t)n * 512 + din) * 3 + tap];
}

// ---------------------------------------------------------------------------
// Embedding: h = x@in_w + in_b + pos_scale*pos + temp_scale*(tf@temp_w + temp_b)
// ---------------------------------------------------------------------------
__global__ __launch_bounds__(256) void embed_kernel(
    const float* __restrict__ x, const float* __restrict__ tf,
    const float* __restrict__ inw, const float* __restrict__ inb,
    const float* __restrict__ pos, const float* __restrict__ tw,
    const float* __restrict__ tb, const float* __restrict__ psc,
    const float* __restrict__ tsc, float* __restrict__ h,
    int M, int D, int IN, int L) {
  const int idx = blockIdx.x * 256 + threadIdx.x;
  if (idx >= M * D) return;
  const int m = idx / D, d = idx - m * D;
  float acc = inb[d];
  const float* xr = x + (size_t)m * IN;
#pragma unroll 8
  for (int i = 0; i < IN; ++i) acc += xr[i] * inw[(size_t)i * D + d];
  const int t = m % L;
  acc += psc[0] * pos[(size_t)t * D + d];
  acc += tsc[0] * (tb[d] + tf[(size_t)m * 2] * tw[d] + tf[(size_t)m * 2 + 1] * tw[D + d]);
  h[idx] = acc;
}

// ---------------------------------------------------------------------------
// LayerNorm over D=512, one block (8 waves) per row.
// ---------------------------------------------------------------------------
__global__ __launch_bounds__(256) void ln_kernel(
    const float* __restrict__ in, const float* __restrict__ s,
    const float* __restrict__ b, float* __restrict__ out) {
  const int row = blockIdx.x, tid = threadIdx.x;
  const float* rp = in + (size_t)row * 512;
  const float v0 = rp[tid], v1 = rp[tid + 256];
  float sum = v0 + v1, sq = v0 * v0 + v1 * v1;
  for (int o = 16; o > 0; o >>= 1) {
    sum += __shfl_xor(sum, o, 32);
    sq  += __shfl_xor(sq,  o, 32);
  }
  __shared__ float rs[8], rq[8];
  if ((tid & 31) == 0) { rs[tid >> 5] = sum; rq[tid >> 5] = sq; }
  __syncthreads();
  sum = 0.f; sq = 0.f;
#pragma unroll
  for (int i = 0; i < 8; ++i) { sum += rs[i]; sq += rq[i]; }
  const float mean = sum * (1.f / 512.f);
  const float var  = sq * (1.f / 512.f) - mean * mean;
  const float r    = rsqrtf(var + 1e-5f);
  float* op = out + (size_t)row * 512;
  op[tid]       = (v0 - mean) * r * s[tid]       + b[tid];
  op[tid + 256] = (v1 - mean) * r * s[tid + 256] + b[tid + 256];
}

// ---------------------------------------------------------------------------
// Top-k(256 of 512) + softmax per score row; one wave per row.
// Exact k-th largest via 32-step radix bit-search on order-preserving keys.
// ---------------------------------------------------------------------------
__device__ inline unsigned fkey(float f) {
  const unsigned u = __float_as_uint(f);
  return (u & 0x80000000u) ? ~u : (u | 0x80000000u);
}

__global__ __launch_bounds__(128) void topk_softmax_kernel(
    float* __restrict__ sc, int nrows, int knum) {
  const int row = blockIdx.x * 4 + (threadIdx.x >> 5);
  if (row >= nrows) return;
  const int lane = threadIdx.x & 31;
  float* rp = sc + (size_t)row * 512;

  float s[16]; unsigned key[16];
#pragma unroll
  for (int i = 0; i < 16; ++i) s[i] = rp[lane + i * 32];

  float mx = -3.4e38f;
#pragma unroll
  for (int i = 0; i < 16; ++i) mx = fmaxf(mx, s[i]);
  for (int o = 16; o > 0; o >>= 1) mx = fmaxf(mx, __shfl_xor(mx, o, 32));

#pragma unroll
  for (int i = 0; i < 16; ++i) key[i] = fkey(s[i]);

  unsigned T = 0;  // largest T with count(key >= T) >= knum == k-th largest key
  for (int bit = 31; bit >= 0; --bit) {
    const unsigned T2 = T | (1u << bit);
    int c = 0;
#pragma unroll
    for (int i = 0; i < 16; ++i) c += (key[i] >= T2) ? 1 : 0;
    for (int o = 16; o > 0; o >>= 1) c += __shfl_xor(c, o, 32);
    if (c >= knum) T = T2;
  }

  float p[16]; float sum = 0.f;
#pragma unroll
  for (int i = 0; i < 16; ++i) {
    p[i] = (key[i] >= T) ? __expf(s[i] - mx) : 0.f;
    sum += p[i];
  }
  for (int o = 16; o > 0; o >>= 1) sum += __shfl_xor(sum, o, 32);
  const float inv = 1.f / sum;
#pragma unroll
  for (int i = 0; i < 16; ++i) rp[lane + i * 32] = p[i] * inv;
}

// ---------------------------------------------------------------------------
// Host orchestration
// ---------------------------------------------------------------------------
static inline GArgs gdefault(int M, int N, int K) {
  GArgs g{};
  g.M = M; g.N = N; g.K = K;
  g.ldb = K;
  g.sectionSize = K; g.shiftBase = 0; g.Lseq = 512;
  g.scale = 1.f; g.relu = 0;
  g.H = 8; g.hd = 64; g.D = 512; g.Lq = 512;
  return g;
}

extern "C" void kernel_launch(void* const* d_in, const int* in_sizes, int n_in,
                              void* d_out, int out_size, void* d_ws, size_t ws_size,
                              hipStream_t stream) {
  (void)in_sizes; (void)n_in; (void)out_size; (void)ws_size;
  constexpr int Bc = 8, Lc = 512, INc = 64, Dc = 512, Hc = 8, Ec = 4, DFc = 2048;
  constexpr int M = Bc * Lc;                  // 4096
  constexpr int KNUM = 256;                   // max(1, int(0.5*L))

  const float* x     = (const float*)d_in[0];
  const float* tf    = (const float*)d_in[1];
  const float* inw   = (const float*)d_in[2];
  const float* inb   = (const float*)d_in[3];
  const float* pos   = (const float*)d_in[4];
  const float* tw    = (const float*)d_in[5];
  const float* tb    = (const float*)d_in[6];
  const float* psc   = (const float*)d_in[7];
  const float* tsc   = (const float*)d_in[8];
  const float* convw = (const float*)d_in[9];
  const float* convb = (const float*)d_in[10];
  const float* qw  = (const float*)d_in[11];
  const float* qbb = (const float*)d_in[12];
  const float* kw  = (const float*)d_in[13];
  const float* kbb = (const float*)d_in[14];
  const float* vw  = (const float*)d_in[15];
  const float* vbb = (const float*)d_in[16];
  const float* ow  = (const float*)d_in[17];
  const float* obb = (const float*)d_in[18];
  const float* f1w = (const float*)d_in[19];
  const float* f1b = (const float*)d_in[20];
  const float* f2w = (const float*)d_in[21];
  const float* f2b = (const float*)d_in[22];
  const float* n1s = (const float*)d_in[23];
  const float* n1b = (const float*)d_in[24];
  const float* n2s = (const float*)d_in[25];
  const float* n2b = (const float*)d_in[26];

  float* h  = (float*)d_out;                  // master activations [4096,512] fp32
  float* ws = (float*)d_ws;
  const size_t MD = (size_t)M * Dc;           // 2,097,152
  float* tmp = ws;                            // 8 MB (conv-out / attn-out)
  float* q_  = ws + MD;                       // 8 MB
  float* k_  = ws + 2 * MD;                   // 8 MB
  float* big = ws + 3 * MD;                   // 64 MB: scores/probs -> preLN2 -> FFN mid
  _Float16* vT     = (_Float16*)(big + (size_t)64 * 512 * 512);   // f16 V^T [B][D][L], 4 MB
  _Float16* wtbase = vT + (size_t)Bc * Dc * Lc;                   // f16 weights, 31.5 MB

  // per-layer f16 weight slabs: conv | q | k | v | o | f1 | f2
  const size_t WCONV = (size_t)Dc * 3 * Dc;       // 786432
  const size_t WSQ   = (size_t)Dc * Dc;           // 262144
  const size_t WF    = (size_t)Dc * DFc;          // 1048576
  const size_t WLAYER = WCONV + 4 * WSQ + 2 * WF; // 3,932,160
  const size_t DD = (size_t)Dc * Dc;

  // ---- one-time weight conversion (re-done each call; deterministic) ----
  for (int l = 0; l < Ec; ++l) {
    _Float16* wl = wtbase + (size_t)l * WLAYER;
    _Float16* wtc  = wl;
    _Float16* wtq  = wl + WCONV;
    _Float16* wtk  = wtq + WSQ;
    _Float16* wtv  = wtk + WSQ;
    _Float16* wto  = wtv + WSQ;
    _Float16* wtf1 = wto + WSQ;
    _Float16* wtf2 = wtf1 + WF;
    conv_wt_kernel<<<(512 * 1536 + 255) / 256, 256, 0, stream>>>(convw + (size_t)l * DD * 3, wtc);
    transpose_f16_kernel<<<dim3(Dc / 32, Dc / 32), 256, 0, stream>>>(qw + l * DD, wtq, Dc, Dc);
    transpose_f16_kernel<<<dim3(Dc / 32, Dc / 32), 256, 0, stream>>>(kw + l * DD, wtk, Dc, Dc);
    transpose_f16_kernel<<<dim3(Dc / 32, Dc / 32), 256, 0, stream>>>(vw + l * DD, wtv, Dc, Dc);
    transpose_f16_kernel<<<dim3(Dc / 32, Dc / 32), 256, 0, stream>>>(ow + l * DD, wto, Dc, Dc);
    transpose_f16_kernel<<<dim3(DFc / 32, Dc / 32), 256, 0, stream>>>(f1w + (size_t)l * WF, wtf1, Dc, DFc);
    transpose_f16_kernel<<<dim3(Dc / 32, DFc / 32), 256, 0, stream>>>(f2w + (size_t)l * WF, wtf2, DFc, Dc);
  }

  embed_kernel<<<(M * Dc + 255) / 256, 256, 0, stream>>>(
      x, tf, inw, inb, pos, tw, tb, psc, tsc, h, M, Dc, INc, Lc);

  for (int l = 0; l < Ec; ++l) {
    _Float16* wl = wtbase + (size_t)l * WLAYER;
    _Float16* wtc  = wl;
    _Float16* wtq  = wl + WCONV;
    _Float16* wtk  = wtq + WSQ;
    _Float16* wtv  = wtk + WSQ;
    _Float16* wto  = wtv + WSQ;
    _Float16* wtf1 = wto + WSQ;
    _Float16* wtf2 = wtf1 + WF;

    // --- causal conv (GEMM over K=3D with shifted taps) + h residual ---
    {
      GArgs g = gdefault(M, Dc, 3 * Dc);
      g.A = h; g.lda = Dc;
      g.Bh = wtc; g.ldb = 3 * Dc;
      g.bias = convb + (size_t)l * Dc; g.resid = h;
      g.C = tmp; g.ldc = Dc;
      g.sectionSize = Dc; g.shiftBase = -2;
      gemm_wmma<MODE_CONV><<<dim3(Dc / 128, M / 64, 1), 128, 0, stream>>>(g);
    }
    ln_kernel<<<M, 256, 0, stream>>>(tmp, n1s + (size_t)l * Dc, n1b + (size_t)l * Dc, h);

    // --- Q, K projections (fp32 out); V projection -> f16 V^T ---
    {
      GArgs g = gdefault(M, Dc, Dc);
      g.A = h; g.lda = Dc; g.Bh = wtq; g.bias = qbb + (size_t)l * Dc;
      g.C = q_; g.ldc = Dc;
      gemm_wmma<MODE_LIN><<<dim3(Dc / 128, M / 64, 1), 128, 0, stream>>>(g);
      g.Bh = wtk; g.bias = kbb + (size_t)l * Dc; g.C = k_;
      gemm_wmma<MODE_LIN><<<dim3(Dc / 128, M / 64, 1), 128, 0, stream>>>(g);
    }
    {
      GArgs g = gdefault(M, Dc, Dc);
      g.A = h; g.lda = Dc; g.Bh = wtv; g.bias = vbb + (size_t)l * Dc;
      g.Cv = vT; g.ldc = Dc;
      gemm_wmma<MODE_VPROJ><<<dim3(Dc / 128, M / 64, 1), 128, 0, stream>>>(g);
    }

    // --- scores = scale * Q K^T + causal, per (b,h) ---
    {
      GArgs g = gdefault(Lc, Lc, 64);
      g.A = q_; g.lda = Dc;
      g.Bf = k_;                      // fp32 B operand, contiguous in k
      g.C = big; g.ldc = Lc;
      g.scale = 0.125f;               // 64^-0.5
      gemm_wmma<MODE_SCORES><<<dim3(Lc / 128, Lc / 64, Bc * Hc), 128, 0, stream>>>(g);
    }

    // --- top-k(256) + softmax, in place ---
    topk_softmax_kernel<<<(Bc * Hc * Lc) / 4, 128, 0, stream>>>(big, Bc * Hc * Lc, KNUM);

    // --- attn_out = P @ V (B operand = f16 V^T, contiguous) ---
    {
      GArgs g = gdefault(Lc, 64, Lc);
      g.A = big; g.lda = Lc;
      g.Bh = vT;
      g.C = tmp; g.ldc = Dc;
      gemm_wmma<MODE_PV><<<dim3(1, Lc / 64, Bc * Hc), 128, 0, stream>>>(g);
    }

    // --- output projection + residual, then LN2 ---
    {
      GArgs g = gdefault(M, Dc, Dc);
      g.A = tmp; g.lda = Dc; g.Bh = wto;
      g.bias = obb + (size_t)l * Dc; g.resid = h;
      g.C = big; g.ldc = Dc;          // scores region dead now
      gemm_wmma<MODE_LIN><<<dim3(Dc / 128, M / 64, 1), 128, 0, stream>>>(g);
    }
    ln_kernel<<<M, 256, 0, stream>>>(big, n2s + (size_t)l * Dc, n2b + (size_t)l * Dc, h);

    // --- FFN: mid = relu(h@f1w + f1b); h += mid@f2w + f2b ---
    {
      GArgs g = gdefault(M, DFc, Dc);
      g.A = h; g.lda = Dc; g.Bh = wtf1;
      g.bias = f1b + (size_t)l * DFc; g.relu = 1;
      g.C = big; g.ldc = DFc;
      gemm_wmma<MODE_LIN><<<dim3(DFc / 128, M / 64, 1), 128, 0, stream>>>(g);
    }
    {
      GArgs g = gdefault(M, Dc, DFc);
      g.A = big; g.lda = DFc; g.Bh = wtf2;
      g.bias = f2b + (size_t)l * Dc; g.resid = h;
      g.C = h; g.ldc = Dc;            // in-place residual: same thread reads+writes
      gemm_wmma<MODE_LIN><<<dim3(Dc / 128, M / 64, 1), 128, 0, stream>>>(g);
    }
  }
}
